// KANFeatureExtractor_3788161155155
// MI455X (gfx1250) — matrix operand order repaired and checked by reading, not compiled, not run on hardware
//
#include <hip/hip_runtime.h>
#include <cstdint>

// CDNA5 wave32 WMMA types (bridge-doc-confirmed signatures)
typedef _Float16 v16h __attribute__((ext_vector_type(16)));
typedef _Float16 v8h  __attribute__((ext_vector_type(8)));
typedef float    v8f  __attribute__((ext_vector_type(8)));

// ---------------------------------------------------------------------------
// KAN pieces: silu and cubic B-spline basis on the fixed knot grid
// t[i] = -1 + (i-3)*0.4, i = 0..11  (GRID_SIZE=5, SPLINE_ORDER=3, N_BASIS=8)
// ---------------------------------------------------------------------------
__device__ __forceinline__ float silu_f(float x) {
  return x / (1.0f + __expf(-x));
}

__device__ __forceinline__ void bspline8(float xv, float bs[8]) {
  float t[12];
#pragma unroll
  for (int i = 0; i < 12; ++i) t[i] = -2.2f + 0.4f * (float)i;
  float b[11];
#pragma unroll
  for (int j = 0; j < 11; ++j)
    b[j] = (xv >= t[j] && xv < t[j + 1]) ? 1.0f : 0.0f;
#pragma unroll
  for (int k = 1; k <= 3; ++k) {
#pragma unroll
    for (int j = 0; j < 11 - k; ++j) {
      float l = (xv - t[j])       / (t[j + k]     - t[j]);
      float r = (t[j + k + 1] - xv) / (t[j + k + 1] - t[j + 1]);
      b[j] = l * b[j] + r * b[j + 1];
    }
  }
#pragma unroll
  for (int j = 0; j < 8; ++j) bs[j] = b[j];
}

// ---------------------------------------------------------------------------
// Fused KAN-conv (3x3, pad 1) + 2x2 maxpool layer.
//   CIN : input channels (1 or 2), COUT fixed at 2
//   H   : input spatial size (H x H), conv output also H x H
//   T   : number of 4x4 conv tiles per dim (ceil(H/4))
//   PH  : pooled output size (H/2)
// One wave handles one 4x4 conv tile -> 2x2 pooled outputs.
// Feature vector per conv pixel: K = CIN*9*9  (per (ci,tap): [silu, basis0..7])
// GEMM: A[16 x K] features (f16, via LDS), B[K x 16] weights (N<2 live),
//       D[16 x 16] f32 accumulators -> in-register 2x2 maxpool -> store.
// ---------------------------------------------------------------------------
template <int CIN, int H, int T, int PH>
__global__ __launch_bounds__(256) void kan_layer_kernel(
    const float* __restrict__ in,        // (B, CIN, H, H)
    const float* __restrict__ base_w,    // (2, CIN, 9)
    const float* __restrict__ spline_w,  // (2, CIN, 9, 8)
    const float* __restrict__ scaler,    // (2, CIN, 9)
    float* __restrict__ out,             // (B, 2, PH, PH)
    int total_tiles)                     // B * T * T
{
  constexpr int NP     = CIN * 9;            // (ci,tap) pairs
  constexpr int KTOT   = NP * 9;             // 81 or 162
  constexpr int KPAD   = (KTOT + 31) / 32 * 32;  // 96 or 192
  constexpr int NKB    = KPAD / 32;          // WMMA K-blocks: 3 or 6
  constexpr int KSTR   = KPAD + 8;           // LDS row stride (halves), 16B-multiple

  __shared__ __align__(16) _Float16 feat[8][16][KSTR];

  const int lane = threadIdx.x & 31;
  const int wv   = threadIdx.x >> 5;
  int wid = blockIdx.x * 8 + wv;
  const bool valid = (wid < total_tiles);
  if (!valid) wid = 0;                      // keep EXEC full; suppress stores

  const int TT  = T * T;
  const int b   = wid / TT;
  const int rem = wid % TT;
  const int y0  = (rem / T) * 4;
  const int x0  = (rem % T) * 4;

  const int row = lane & 15;                // conv pixel within tile (M row)
  const int g   = lane >> 4;                // half-wave selector

  // ---- B fragments: weights as K x N (N = lane&15), halves j -> K=g*16+j ---
  const int n = lane & 15;
  v16h bfrag[NKB];
#pragma unroll
  for (int kb = 0; kb < NKB; ++kb) {
#pragma unroll
    for (int j = 0; j < 16; ++j) {
      const int K = kb * 32 + g * 16 + j;
      float wvv = 0.0f;
      if (n < 2 && K < KTOT) {
        const int p   = K / 9;              // ci*9 + tap
        const int f   = K % 9;              // 0 = silu, 1..8 = basis
        const int ci  = p / 9;
        const int tap = p % 9;
        const int widx = (n * CIN + ci) * 9 + tap;
        if (f == 0) wvv = base_w[widx];
        else        wvv = spline_w[widx * 8 + (f - 1)] * scaler[widx];
      }
      bfrag[kb][j] = (_Float16)wvv;
    }
  }

  // ---- compute f16 features into LDS (this wave's private tile) -----------
  const int cy = y0 + (row >> 2);
  const int cx = x0 + (row & 3);
  const float* inb = in + (size_t)b * CIN * H * H;
  const int pstart = g ? (NP + 1) / 2 : 0;
  const int pend   = g ? NP           : (NP + 1) / 2;
  for (int p = pstart; p < pend; ++p) {
    const int ci  = p / 9;
    const int tap = p % 9;
    const int iy  = cy + tap / 3 - 1;
    const int ix  = cx + tap % 3 - 1;
    float v = 0.0f;
    if (iy >= 0 && iy < H && ix >= 0 && ix < H)
      v = inb[ci * H * H + iy * H + ix];
    float bs[8];
    bspline8(v, bs);
    _Float16* fr = &feat[wv][row][p * 9];
    fr[0] = (_Float16)silu_f(v);
#pragma unroll
    for (int j = 0; j < 8; ++j) fr[1 + j] = (_Float16)bs[j];
  }
  if (g) {  // zero the K padding [KTOT, KPAD)
    for (int k = KTOT; k < KPAD; ++k) feat[wv][row][k] = (_Float16)0.0f;
  }
  __syncthreads();

  // ---- WMMA: D[16x16] += A[16x32] * B[32x16] over NKB K-blocks ------------
  v8f acc = {0.f, 0.f, 0.f, 0.f, 0.f, 0.f, 0.f, 0.f};
  const _Float16* rp = &feat[wv][row][0];
#pragma unroll
  for (int kb = 0; kb < NKB; ++kb) {
    // 16-bit A layout: lane halves 0..15 = K kb*32+g*16.. (contiguous, matches B)
    v8h lo = *(const v8h*)(rp + kb * 32 + g * 16);
    v8h hi = *(const v8h*)(rp + kb * 32 + g * 16 + 8);
    v16h a;
#pragma unroll
    for (int j = 0; j < 8; ++j) { a[j] = lo[j]; a[8 + j] = hi[j]; }
    acc = __builtin_amdgcn_wmma_f32_16x16x32_f16(
        /*neg_a=*/false, a, /*neg_b=*/false, bfrag[kb],
        /*c_mod=*/(short)0, acc, /*reuse_a=*/false, /*reuse_b=*/false);
  }

  // ---- in-register 2x2 maxpool + store ------------------------------------
  // D: VGPR r, lane L -> conv row m = r + 8*g, channel n = L&15.
  // m -> (my=m>>2, mx=m&3); pooled py = g, px from VGPR groups.
  const float p0 = fmaxf(fmaxf(acc[0], acc[1]), fmaxf(acc[4], acc[5])); // px=0
  const float p1 = fmaxf(fmaxf(acc[2], acc[3]), fmaxf(acc[6], acc[7])); // px=1
  if (valid && n < 2) {
    const int gy = (y0 >> 1) + g;
    const int gx = (x0 >> 1);
    float* ob = out + ((size_t)b * 2 + n) * PH * PH;
    if (gy < PH) {
      if (gx     < PH) ob[gy * PH + gx]     = p0;
      if (gx + 1 < PH) ob[gy * PH + gx + 1] = p1;
    }
  }
}

// ---------------------------------------------------------------------------
// Launch: layer1 (1->2ch, 28x28 -> pooled 14x14 in d_ws),
//         layer2 (2->2ch, 14x14 -> pooled 7x7 in d_out)
// ---------------------------------------------------------------------------
extern "C" void kernel_launch(void* const* d_in, const int* in_sizes, int n_in,
                              void* d_out, int out_size, void* d_ws, size_t ws_size,
                              hipStream_t stream) {
  const float* x        = (const float*)d_in[0];
  const float* base_w1  = (const float*)d_in[1];
  const float* spline_w1= (const float*)d_in[2];
  const float* scaler1  = (const float*)d_in[3];
  const float* base_w2  = (const float*)d_in[4];
  const float* spline_w2= (const float*)d_in[5];
  const float* scaler2  = (const float*)d_in[6];

  const int B = in_sizes[0] / (28 * 28);   // Cin=1
  float* pooled1 = (float*)d_ws;           // (B, 2, 14, 14)

  {
    const int total = B * 7 * 7;           // 4x4 conv tiles over 28x28
    const int blocks = (total + 7) / 8;    // 8 waves per 256-thread block
    kan_layer_kernel<1, 28, 7, 14><<<blocks, 256, 0, stream>>>(
        x, base_w1, spline_w1, scaler1, pooled1, total);
  }
  {
    const int total = B * 4 * 4;           // 4x4 conv tiles over 14x14
    const int blocks = (total + 7) / 8;
    kan_layer_kernel<2, 14, 4, 7><<<blocks, 256, 0, stream>>>(
        pooled1, base_w2, spline_w2, scaler2, (float*)d_out, total);
  }
}
